// Attention_46471546142816
// MI455X (gfx1250) — matrix operand order repaired and checked by reading, not compiled, not run on hardware
//
#include <hip/hip_runtime.h>
#include <cstdint>
#include <cstddef>

// ---------------------------------------------------------------------------
// Types for CDNA5 WMMA (wave32): v_wmma_f32_16x16x32_bf16
// ---------------------------------------------------------------------------
typedef __bf16 bf16_t;
typedef bf16_t v16bf __attribute__((ext_vector_type(16)));
typedef float  v8f   __attribute__((ext_vector_type(8)));
typedef unsigned int u32x4 __attribute__((ext_vector_type(4)));
typedef int i32x4 __attribute__((ext_vector_type(4)));
typedef int i32x8 __attribute__((ext_vector_type(8)));

#define WMMA_BF16(a, b, c) \
  __builtin_amdgcn_wmma_f32_16x16x32_bf16(false, (a), false, (b), (short)0, (c), false, false)

__device__ __forceinline__ bf16_t f2bf(float f) {
  unsigned u = __builtin_bit_cast(unsigned, f);
  unsigned r = u + 0x7FFFu + ((u >> 16) & 1u);   // round-to-nearest-even
  unsigned short h = (unsigned short)(r >> 16);
  return __builtin_bit_cast(bf16_t, h);
}

union FragU { v16bf v; u32x4 q[2]; };

// A-matrix 16x32 bf16 fragment (ISA 7.12.2): lane row = l16, elements cover
// K = kbase + 8*hi + {0..7} and K = kbase + 8*hi + 16 + {0..7}.
__device__ __forceinline__ v16bf load_a_frag(const bf16_t* base, int stride,
                                             int l16, int hi, int kbase) {
  const bf16_t* r = base + l16 * stride + kbase + 8 * hi;
  FragU u;
  u.q[0] = *(const u32x4*)(r);
  u.q[1] = *(const u32x4*)(r + 16);
  return u.v;
}

// B-matrix 32x16 bf16 fragment: lane col = l16 (a row in the K-contiguous LDS
// tile), elements cover K = kbase + 8*hi + {0..7} and K = kbase + 8*hi + 16 + {0..7}.
__device__ __forceinline__ v16bf load_b_frag(const bf16_t* base, int stride,
                                             int l16, int hi, int kbase) {
  const bf16_t* r = base + l16 * stride + kbase + 8 * hi;
  FragU u;
  u.q[0] = *(const u32x4*)(r);
  u.q[1] = *(const u32x4*)(r + 16);
  return u.v;
}

// ---------------------------------------------------------------------------
// TDM: DMA one [128 rows x 32 bf16] tile (row stride = row_stride_elems in
// global) into LDS with a 40-element (80 B) padded LDS row stride.
// D# layout per CDNA5 ISA 8.3-8.6:
//   g0: count=1 | lds_addr | global_addr(57b) | type=2
//   g1: data_size=1(2B), pad_enable, pad_interval=3(16 DW = 64 B rows),
//       pad_amount=3(4 DW = 8 elems); tensor_dim0=32, tensor_dim1=128,
//       tile_dim0=32, tile_dim1=128, tensor_dim0_stride=row_stride_elems
// 6-arg builtin form (amdgpu-toolchain clang-23): extra int32x8 zero-filled.
// ---------------------------------------------------------------------------
__device__ __forceinline__ void tdm_load_tile_128x32(const bf16_t* gsrc,
                                                     unsigned lds_off,
                                                     int row_stride_elems) {
  unsigned long long ga = (unsigned long long)(uintptr_t)gsrc;
  u32x4 g0;
  g0[0] = 1u;                                            // count=1 (valid), user
  g0[1] = lds_off;                                       // lds_addr (bytes)
  g0[2] = (unsigned)ga;                                  // global_addr[31:0]
  g0[3] = ((unsigned)(ga >> 32) & 0x01FFFFFFu) | 0x80000000u;  // addr[56:32]|type=2
  i32x8 g1;
  g1[0] = (int)((1u << 16) | (1u << 20) | (3u << 22) | (3u << 25));
  g1[1] = (int)(32u << 16);                              // tensor_dim0 = 32 (lo16)
  g1[2] = (int)(128u << 16);                             // dim0 hi | tensor_dim1 = 128
  g1[3] = (int)(32u << 16);                              // dim1 hi | tile_dim0 = 32
  g1[4] = 128;                                           // tile_dim1=128, tile_dim2=0
  g1[5] = row_stride_elems;                              // tensor_dim0_stride lo32
  g1[6] = 0;
  g1[7] = 0;
  i32x4 gz4 = {0, 0, 0, 0};
  i32x8 gz8 = {0, 0, 0, 0, 0, 0, 0, 0};
  __builtin_amdgcn_tensor_load_to_lds(g0, g1, gz4, gz4, gz8, 0);
}

// ---------------------------------------------------------------------------
// f32 -> bf16 conversion (straight, vectorized)
// ---------------------------------------------------------------------------
__global__ __launch_bounds__(256) void cvt_bf16_kernel(const float* __restrict__ src,
                                                       bf16_t* __restrict__ dst, int n4) {
  int i = blockIdx.x * 256 + threadIdx.x;
  if (i < n4) {
    float4 v = ((const float4*)src)[i];
    unsigned long long pk =
        (unsigned long long)__builtin_bit_cast(unsigned short, f2bf(v.x)) |
        ((unsigned long long)__builtin_bit_cast(unsigned short, f2bf(v.y)) << 16) |
        ((unsigned long long)__builtin_bit_cast(unsigned short, f2bf(v.z)) << 32) |
        ((unsigned long long)__builtin_bit_cast(unsigned short, f2bf(v.w)) << 48);
    *(unsigned long long*)&dst[(size_t)i * 4] = pk;
  }
}

// f32 W[K][N] -> bf16 Wt[N][K]  (weights are tiny; done once)
__global__ __launch_bounds__(256) void cvt_t_bf16_kernel(const float* __restrict__ W,
                                                         bf16_t* __restrict__ Wt,
                                                         int K, int N) {
  int e = blockIdx.x * 256 + threadIdx.x;   // e over N*K
  int n = e / K, k = e - n * K;
  Wt[e] = f2bf(W[(size_t)k * N + n]);
}

// ---------------------------------------------------------------------------
// GEMM: C[M,N] = A_bf16[M,K] @ B (as Bt_bf16[N,K])
//   mode 0: f32 out + bias (final projection)
//   mode 1: RoPE + scale + pack to (B,H,S,64) bf16         (Q)
//   mode 2: cols<1024 -> RoPE pack (K); cols>=1024 -> V packed transposed (B,H,64,S)
// Tile 128x128, 8 waves (each 32x64 = 2x4 WMMA accums), K-step 32.
// Double-buffered LDS tiles filled by the Tensor Data Mover; one barrier/iter,
// DMA of tile i+1 overlaps the 8 WMMAs of tile i.
// ---------------------------------------------------------------------------
__global__ __launch_bounds__(256) void gemm_bf16_kernel(
    const bf16_t* __restrict__ Ag, const bf16_t* __restrict__ Btg,
    int M, int N, int K, int mode,
    float* __restrict__ outF, const float* __restrict__ bias,
    bf16_t* __restrict__ outP0, bf16_t* __restrict__ outP1,
    const float* __restrict__ cosp, const float* __restrict__ sinp, float oscale) {
  __shared__ __attribute__((aligned(16))) bf16_t Atile[2][128 * 40];
  __shared__ __attribute__((aligned(16))) bf16_t Btile[2][128 * 40];

  const int t    = threadIdx.x;
  const int lane = t & 31, wave = t >> 5;
  const int l16  = lane & 15, hi = lane >> 4;
  const int wm   = wave & 3, wn = wave >> 2;
  const int m0   = blockIdx.y * 128, n0 = blockIdx.x * 128;

  const bf16_t* Abase = Ag + (size_t)m0 * K;
  const bf16_t* Bbase = Btg + (size_t)n0 * K;
  const unsigned ldsA[2] = {(unsigned)(uintptr_t)&Atile[0][0],
                            (unsigned)(uintptr_t)&Atile[1][0]};
  const unsigned ldsB[2] = {(unsigned)(uintptr_t)&Btile[0][0],
                            (unsigned)(uintptr_t)&Btile[1][0]};

  v8f zero = {};
  v8f acc[2][4];
#pragma unroll
  for (int mt = 0; mt < 2; ++mt)
#pragma unroll
    for (int nt = 0; nt < 4; ++nt) acc[mt][nt] = zero;

  if (wave == 0) {                    // DMA tile 0 into buffer 0
    tdm_load_tile_128x32(Abase, ldsA[0], K);
    tdm_load_tile_128x32(Bbase, ldsB[0], K);
  }

  int p = 0;
  for (int k0 = 0; k0 < K; k0 += 32) {
    if (wave == 0) __builtin_amdgcn_s_wait_tensorcnt(0);  // tile k0 landed in LDS
    __syncthreads();  // tile ready for all; all waves done reading buffer p^1
    if (wave == 0 && k0 + 32 < K) {   // DMA tile k0+32 into the other buffer
      tdm_load_tile_128x32(Abase + k0 + 32, ldsA[p ^ 1], K);
      tdm_load_tile_128x32(Bbase + k0 + 32, ldsB[p ^ 1], K);
    }

    const bf16_t* At = &Atile[p][0];
    const bf16_t* Bt = &Btile[p][0];
    v16bf af[2], bfq[4];
#pragma unroll
    for (int mt = 0; mt < 2; ++mt)
      af[mt] = load_a_frag(At + (wm * 32 + mt * 16) * 40, 40, l16, hi, 0);
#pragma unroll
    for (int nt = 0; nt < 4; ++nt)
      bfq[nt] = load_b_frag(Bt + (wn * 64 + nt * 16) * 40, 40, l16, hi, 0);
#pragma unroll
    for (int mt = 0; mt < 2; ++mt)
#pragma unroll
      for (int nt = 0; nt < 4; ++nt)
        acc[mt][nt] = WMMA_BF16(af[mt], bfq[nt], acc[mt][nt]);
    p ^= 1;
  }

  // ---- epilogue ----
  const int ncol0 = n0 + wn * 64;          // head-aligned (multiple of 64)
#pragma unroll
  for (int mt = 0; mt < 2; ++mt) {
#pragma unroll
    for (int i = 0; i < 8; ++i) {
      int m = m0 + wm * 32 + mt * 16 + i + 8 * hi;
      if (mode == 0) {
#pragma unroll
        for (int nt = 0; nt < 4; ++nt) {
          int n = ncol0 + nt * 16 + l16;
          outF[(size_t)m * N + n] = acc[mt][nt][i] + bias[n];
        }
      } else {
        int b  = m >> 11;                  // S = 2048
        int ns = m & 2047;
#pragma unroll
        for (int nt = 0; nt < 4; ++nt) {
          int n = ncol0 + nt * 16 + l16;
          float a = acc[mt][nt][i];
          if (mode == 2 && n >= 1024) {    // V: pack transposed (B,H,64,S)
            int nv = n - 1024;
            int h = nv >> 6, dd = nv & 63;
            outP1[((size_t)(b * 16 + h) * 64 + dd) * 2048 + ns] = f2bf(a);
          } else {                         // Q or K: RoPE; pair is frag nt^2 in-register
            int h = n >> 6, dd = n & 63;
            float pair = acc[mt][nt ^ 2][i];
            float rot  = ((nt & 2) == 0) ? -pair : pair;   // dd<32 -> -x[dd+32]
            float c = cosp[ns * 64 + dd], s = sinp[ns * 64 + dd];
            float v = (a * c + rot * s) * oscale;
            outP0[((size_t)(b * 16 + h) * 2048 + ns) * 64 + dd] = f2bf(v);
          }
        }
      }
    }
  }
}

// ---------------------------------------------------------------------------
// Flash attention: grid (B*H, S/128). 8 waves x 16 query rows; online softmax.
// qp/kp: (B,H,S,64) bf16 (q pre-scaled by 1/8); vt: (B,H,64,S) bf16.
// Output op: (B,S,H*64) bf16 (A-matrix layout for the out projection GEMM).
// ---------------------------------------------------------------------------
__global__ __launch_bounds__(256) void flash_attn_kernel(
    const bf16_t* __restrict__ qp, const bf16_t* __restrict__ kp,
    const bf16_t* __restrict__ vt, bf16_t* __restrict__ op) {
  __shared__ __attribute__((aligned(16))) bf16_t Qs[128 * 72];
  __shared__ __attribute__((aligned(16))) bf16_t Ks[64 * 72];   // [key][dd]
  __shared__ __attribute__((aligned(16))) bf16_t Vs[64 * 72];   // [dd][key]
  __shared__ __attribute__((aligned(16))) bf16_t Ps[8][16 * 72];

  const int t    = threadIdx.x;
  const int lane = t & 31, wave = t >> 5;
  const int l16  = lane & 15, hi = lane >> 4;
  const int bh   = blockIdx.x;
  const int qbase = blockIdx.y * 128;
  const bf16_t* qh = qp + (size_t)bh * 2048 * 64;
  const bf16_t* kh = kp + (size_t)bh * 2048 * 64;
  const bf16_t* vh = vt + (size_t)bh * 64 * 2048;

#pragma unroll
  for (int rep = 0; rep < 4; ++rep) {      // 1024 u32x4 chunks
    int u = t + rep * 256;
    int row = u >> 3, c8 = (u & 7) * 8;
    *(u32x4*)&Qs[row * 72 + c8] = *(const u32x4*)&qh[(size_t)(qbase + row) * 64 + c8];
  }
  __syncthreads();

  v16bf qa[2];
  qa[0] = load_a_frag(&Qs[(wave * 16) * 72], 72, l16, hi, 0);
  qa[1] = load_a_frag(&Qs[(wave * 16) * 72], 72, l16, hi, 32);

  v8f zero = {};
  v8f of[4] = {zero, zero, zero, zero};
  float mrow[8], lrow[8];
#pragma unroll
  for (int i = 0; i < 8; ++i) { mrow[i] = -3.0e38f; lrow[i] = 0.f; }

  for (int kb = 0; kb < 32; ++kb) {
    __syncthreads();
#pragma unroll
    for (int rep = 0; rep < 2; ++rep) {    // 512 chunks each for K and Vt
      int u = t + rep * 256;
      int row = u >> 3, c8 = (u & 7) * 8;
      *(u32x4*)&Ks[row * 72 + c8] = *(const u32x4*)&kh[(size_t)(kb * 64 + row) * 64 + c8];
      *(u32x4*)&Vs[row * 72 + c8] = *(const u32x4*)&vh[(size_t)row * 2048 + kb * 64 + c8];
    }
    if (kb + 1 < 32) {                     // warm caches for next key block
      int r = t >> 2;                      // 0..63
      __builtin_prefetch(&kh[(size_t)((kb + 1) * 64 + r) * 64], 0, 1);
      __builtin_prefetch(&vh[(size_t)r * 2048 + (kb + 1) * 64], 0, 1);
    }
    __syncthreads();

    // S (16 queries x 64 keys); scale already folded into q
    v8f sf[4];
#pragma unroll
    for (int nf = 0; nf < 4; ++nf) {
      v8f a = zero;
#pragma unroll
      for (int ks = 0; ks < 2; ++ks) {
        v16bf kbf = load_b_frag(&Ks[(nf * 16) * 72], 72, l16, hi, ks * 32);
        a = WMMA_BF16(qa[ks], kbf, a);
      }
      sf[nf] = a;
    }

    // online softmax (rows live across a 16-lane half under wave32)
    float alpha[8];
#pragma unroll
    for (int i = 0; i < 8; ++i) {
      float mx = fmaxf(fmaxf(sf[0][i], sf[1][i]), fmaxf(sf[2][i], sf[3][i]));
#pragma unroll
      for (int msk = 1; msk < 16; msk <<= 1) mx = fmaxf(mx, __shfl_xor(mx, msk, 32));
      float mn = fmaxf(mrow[i], mx);
      alpha[i] = __expf(mrow[i] - mn);
      mrow[i] = mn;
    }
    bf16_t* pw = &Ps[wave][0];
    float rs[8];
#pragma unroll
    for (int i = 0; i < 8; ++i) rs[i] = 0.f;
#pragma unroll
    for (int nf = 0; nf < 4; ++nf)
#pragma unroll
      for (int i = 0; i < 8; ++i) {
        float p = __expf(sf[nf][i] - mrow[i]);
        rs[i] += p;
        pw[(i + 8 * hi) * 72 + nf * 16 + l16] = f2bf(p);  // C-layout -> row-major LDS
      }
#pragma unroll
    for (int i = 0; i < 8; ++i) {
      float s = rs[i];
#pragma unroll
      for (int msk = 1; msk < 16; msk <<= 1) s += __shfl_xor(s, msk, 32);
      lrow[i] = lrow[i] * alpha[i] + s;
#pragma unroll
      for (int nf = 0; nf < 4; ++nf) of[nf][i] = of[nf][i] * alpha[i];
    }
    asm volatile("s_wait_dscnt 0" ::: "memory");   // wave-private P store->load

    // O += P @ V
#pragma unroll
    for (int ks = 0; ks < 2; ++ks) {
      v16bf pa = load_a_frag(pw, 72, l16, hi, ks * 32);
#pragma unroll
      for (int nf = 0; nf < 4; ++nf) {
        v16bf vb = load_b_frag(&Vs[(nf * 16) * 72], 72, l16, hi, ks * 32);
        of[nf] = WMMA_BF16(pa, vb, of[nf]);
      }
    }
  }

  const int b = bh >> 4, h = bh & 15;
#pragma unroll
  for (int i = 0; i < 8; ++i) {
    float inv = 1.f / lrow[i];
    int n = qbase + wave * 16 + i + 8 * hi;
#pragma unroll
    for (int nf = 0; nf < 4; ++nf) {
      int dd = nf * 16 + l16;
      op[((size_t)(b * 2048 + n)) * 1024 + h * 64 + dd] = f2bf(of[nf][i] * inv);
    }
  }
}

// ---------------------------------------------------------------------------
extern "C" void kernel_launch(void* const* d_in, const int* in_sizes, int n_in,
                              void* d_out, int out_size, void* d_ws, size_t ws_size,
                              hipStream_t stream) {
  const float* x    = (const float*)d_in[0];
  const float* cosp = (const float*)d_in[1];
  const float* sinp = (const float*)d_in[2];
  const float* Wq   = (const float*)d_in[3];
  const float* Wkv  = (const float*)d_in[4];
  const float* Wout = (const float*)d_in[5];
  const float* bout = (const float*)d_in[6];
  float* out = (float*)d_out;
  (void)in_sizes; (void)n_in; (void)out_size; (void)ws_size;

  const int S = 2048, C = 1024, M = 4 * S;   // B=4, H=16, d=64

  char* ws = (char*)d_ws;
  size_t off = 0;
  auto take = [&](size_t bytes) -> char* {
    char* p = ws + off;
    off += (bytes + 255) & ~(size_t)255;
    return p;
  };
  bf16_t* xbf   = (bf16_t*)take((size_t)M * C * 2);      // x bf16
  bf16_t* WqT   = (bf16_t*)take((size_t)C * C * 2);      // Wq^T bf16
  bf16_t* WkvT  = (bf16_t*)take((size_t)2 * C * C * 2);  // Wkv^T bf16
  bf16_t* WoutT = (bf16_t*)take((size_t)C * C * 2);      // Wout^T bf16
  bf16_t* qpk   = (bf16_t*)take((size_t)M * C * 2);      // q packed (B,H,S,64)
  bf16_t* kpk   = (bf16_t*)take((size_t)M * C * 2);      // k packed (B,H,S,64)
  bf16_t* vtr   = (bf16_t*)take((size_t)M * C * 2);      // v packed (B,H,64,S)
  bf16_t* obf   = (bf16_t*)take((size_t)M * C * 2);      // attn out (B,S,C)

  cvt_bf16_kernel<<<(M * C / 4 + 255) / 256, 256, 0, stream>>>(x, xbf, M * C / 4);
  cvt_t_bf16_kernel<<<(C * C) / 256, 256, 0, stream>>>(Wq, WqT, C, C);
  cvt_t_bf16_kernel<<<(2 * C * C) / 256, 256, 0, stream>>>(Wkv, WkvT, C, 2 * C);
  cvt_t_bf16_kernel<<<(C * C) / 256, 256, 0, stream>>>(Wout, WoutT, C, C);

  // Q = x@Wq, RoPE, scale d^-0.5 = 1/8, pack per-head
  gemm_bf16_kernel<<<dim3(C / 128, M / 128), 256, 0, stream>>>(
      xbf, WqT, M, C, C, 1, nullptr, nullptr, qpk, nullptr, cosp, sinp, 0.125f);
  // KV = x@Wkv; K gets RoPE, V packed transposed
  gemm_bf16_kernel<<<dim3(2 * C / 128, M / 128), 256, 0, stream>>>(
      xbf, WkvT, M, 2 * C, C, 2, nullptr, nullptr, kpk, vtr, cosp, sinp, 1.0f);
  // softmax(QK^T)V
  flash_attn_kernel<<<dim3(64, S / 128), 256, 0, stream>>>(qpk, kpk, vtr, obf);
  // out = O@Wout + bout  (f32)
  gemm_bf16_kernel<<<dim3(C / 128, M / 128), 256, 0, stream>>>(
      obf, WoutT, M, C, C, 0, out, bout, nullptr, nullptr, nullptr, nullptr, 1.0f);
}
